// GraphAttentionNet_33938831573034
// MI455X (gfx1250) — matrix operand order repaired and checked by reading, not compiled, not run on hardware
//
#include <hip/hip_runtime.h>
#include <hip/hip_bf16.h>
#include <stdint.h>

typedef __attribute__((ext_vector_type(16))) _Float16 v16h;
typedef __attribute__((ext_vector_type(8)))  float    v8f;

#define HDIM   256
#define NHEADS 8

// ---------------------------------------------------------------- helpers

__device__ __forceinline__ float waveSum(float v) {
#pragma unroll
  for (int m = 16; m; m >>= 1) v += __shfl_xor(v, m, 32);
  return v;
}

__device__ __forceinline__ float blockReduceSum(float v, float* sm) {
  __syncthreads();
  int lane = threadIdx.x & 31, w = threadIdx.x >> 5;
  v = waveSum(v);
  if (lane == 0) sm[w] = v;
  __syncthreads();
  if (w == 0) {
    int nw = (blockDim.x + 31) >> 5;
    float r = (lane < nw) ? sm[lane] : 0.f;
    r = waveSum(r);
    if (lane == 0) sm[0] = r;
  }
  __syncthreads();
  return sm[0];
}

// monotonic float->uint encoding for atomicMax on floats of any sign
__device__ __forceinline__ void atomicMaxF(unsigned* addr, float v) {
  unsigned uv = __float_as_uint(v);
  unsigned key = (uv & 0x80000000u) ? ~uv : (uv | 0x80000000u);
  atomicMax(addr, key);
}
__device__ __forceinline__ float decodeMaxF(unsigned key) {
  return (key & 0x80000000u) ? __uint_as_float(key & 0x7fffffffu)
                             : __uint_as_float(~key);
}

// --------------------------------------------------- WMMA fragment loaders
// A: 16x32 f16 (MxK). lane<16: halves0-7 K=k0..+7, halves8-15 K=k0+16..+23;
// lane>=16 shifted by 8.  M = lane%16.
__device__ __forceinline__ v16h load_A_f32(const float* __restrict__ A, int lda,
                                           int m0, int k0, int lane) {
  const float* p = A + (size_t)(m0 + (lane & 15)) * lda + k0 + ((lane >> 4) << 3);
  v16h a;
#pragma unroll
  for (int q = 0; q < 8; ++q) a[q] = (_Float16)p[q];
#pragma unroll
  for (int q = 0; q < 8; ++q) a[8 + q] = (_Float16)p[16 + q];
  return a;
}
__device__ __forceinline__ v16h load_A_lds(const _Float16* t, int m0, int k0, int lane) {
  const _Float16* p = t + (size_t)(m0 + (lane & 15)) * HDIM + k0 + ((lane >> 4) << 3);
  v16h a;
#pragma unroll
  for (int q = 0; q < 8; ++q) a[q] = p[q];
#pragma unroll
  for (int q = 0; q < 8; ++q) a[8 + q] = p[16 + q];
  return a;
}
// B: 32x16 f16 (KxN). N = lane%16; K in same order as A (k0+8h+s, k0+16+8h+s-8).
__device__ __forceinline__ v16h load_B_f32(const float* __restrict__ B, int ldb,
                                           int k0, int n0, int lane) {
  const float* p = B + (size_t)(k0 + ((lane >> 4) << 3)) * ldb + n0 + (lane & 15);
  v16h b;
#pragma unroll
  for (int q = 0; q < 8; ++q) b[q] = (_Float16)p[(size_t)q * ldb];
#pragma unroll
  for (int q = 0; q < 8; ++q) b[8 + q] = (_Float16)p[(size_t)(16 + q) * ldb];
  return b;
}

// -------------------------------------------------------------- kernels

// generic C = A[M,K] x B[K,N] (+bias), f16 WMMA math, f32 accum/out.
__global__ void k_gemm_wmma(const float* __restrict__ A, const float* __restrict__ B,
                            const float* __restrict__ bias, float* __restrict__ C,
                            int M, int K, int Nn) {
  int lane = threadIdx.x & 31;
  int wave = blockIdx.x * (blockDim.x >> 5) + (threadIdx.x >> 5);
  int ntiles = Nn >> 4;
  int mt = wave / ntiles, nt = wave % ntiles;
  if (mt * 16 >= M) return;                 // wave-uniform
  int m0 = mt * 16, n0 = nt * 16;
  v8f acc = {};
  for (int k0 = 0; k0 < K; k0 += 32) {
    v16h a = load_A_f32(A, K, m0, k0, lane);
    v16h b = load_B_f32(B, Nn, k0, n0, lane);
    acc = __builtin_amdgcn_wmma_f32_16x16x32_f16(false, a, false, b,
                                                 (short)0, acc, false, false);
  }
  int n = n0 + (lane & 15);
  float bv = bias ? bias[n] : 0.f;
#pragma unroll
  for (int r = 0; r < 8; ++r) {
    int m = m0 + r + ((lane >> 4) << 3);
    C[(size_t)m * Nn + n] = acc[r] + bv;
  }
}

__global__ void k_selfsum(const int* __restrict__ src, const int* __restrict__ dst,
                          const float* __restrict__ ea, float* __restrict__ sums,
                          float* __restrict__ cnt, int E) {
  int e = blockIdx.x * blockDim.x + threadIdx.x;
  if (e >= E) return;
  int d = dst[e];
  atomicAdd(&sums[d * 3 + 0], ea[(size_t)e * 3 + 0]);
  atomicAdd(&sums[d * 3 + 1], ea[(size_t)e * 3 + 1]);
  atomicAdd(&sums[d * 3 + 2], ea[(size_t)e * 3 + 2]);
  atomicAdd(&cnt[d], 1.f);
}

__global__ void k_selfattr(const float* __restrict__ sums, const float* __restrict__ cnt,
                           float* __restrict__ sa, int N) {
  int t = blockIdx.x * blockDim.x + threadIdx.x;
  if (t >= N * 3) return;
  float c = cnt[t / 3];
  sa[t] = sums[t] / (c > 1.f ? c : 1.f);
}

__global__ void k_h0(const float* __restrict__ x, const float* __restrict__ W,
                     const float* __restrict__ b, float* __restrict__ h) {
  int n = blockIdx.x, c = threadIdx.x;
  h[(size_t)n * HDIM + c] = x[n] * W[c] + b[c];
}

// wave-per-edge GATv2 score: alpha[e,h] = sum_c leaky(xl[s]+xr[d]+ee)*att
__global__ void k_alpha(const float* __restrict__ xl, const float* __restrict__ xr,
                        const int* __restrict__ src, const int* __restrict__ dst,
                        const float* __restrict__ edge_attr, const float* __restrict__ self_attr,
                        const float* __restrict__ We, const float* __restrict__ att,
                        float* __restrict__ alpha, unsigned* __restrict__ amax,
                        int E, int N) {
  int lane = threadIdx.x & 31;
  long long e = (long long)blockIdx.x * (blockDim.x >> 5) + (threadIdx.x >> 5);
  if (e >= (long long)E + N) return;        // wave-uniform
  int s, d; const float* ea;
  if (e < E) { s = src[e]; d = dst[e]; ea = edge_attr + e * 3; }
  else       { s = d = (int)(e - E);   ea = self_attr + (e - E) * 3; }
  float e0 = ea[0], e1 = ea[1], e2 = ea[2];
  const float* xls = xl + (size_t)s * HDIM;
  const float* xrd = xr + (size_t)d * HDIM;
  float part = 0.f;
#pragma unroll
  for (int q = 0; q < 8; ++q) {
    int ch = lane * 8 + q;
    float m = xls[ch] + xrd[ch] + (We[ch] * e0 + We[HDIM + ch] * e1 + We[2 * HDIM + ch] * e2);
    m = m > 0.f ? m : 0.2f * m;
    part += m * att[ch];                    // att row-major [8][32] == [256]
  }
  part += __shfl_xor(part, 1, 32);
  part += __shfl_xor(part, 2, 32);          // sum over the 4 lanes of a head
  int head = lane >> 2;
  if ((lane & 3) == 0) {
    alpha[e * NHEADS + head] = part;
    atomicMaxF(&amax[d * NHEADS + head], part);
  }
}

__global__ void k_expsum(float* __restrict__ alpha, const unsigned* __restrict__ amax,
                         float* __restrict__ denom, const int* __restrict__ dst,
                         int E, int N) {
  long long t = (long long)blockIdx.x * blockDim.x + threadIdx.x;
  if (t >= ((long long)E + N) * NHEADS) return;
  long long e = t >> 3; int hh = (int)(t & 7);
  int d = (e < E) ? dst[e] : (int)(e - E);
  float v = __expf(alpha[t] - decodeMaxF(amax[d * NHEADS + hh]));
  alpha[t] = v;
  atomicAdd(&denom[d * NHEADS + hh], v);
}

__global__ void k_agg(const float* __restrict__ xl, const float* __restrict__ alpha,
                      const float* __restrict__ denom, const int* __restrict__ src,
                      const int* __restrict__ dst, float* __restrict__ out,
                      int E, int N) {
  int lane = threadIdx.x & 31;
  long long e = (long long)blockIdx.x * (blockDim.x >> 5) + (threadIdx.x >> 5);
  if (e >= (long long)E + N) return;
  int s, d;
  if (e < E) { s = src[e]; d = dst[e]; } else { s = d = (int)(e - E); }
  int head = lane >> 2;
  float w = alpha[e * NHEADS + head] / denom[d * NHEADS + head];
  const float* xls = xl + (size_t)s * HDIM;
  float* od = out + (size_t)d * HDIM;
#pragma unroll
  for (int q = 0; q < 8; ++q) {
    int ch = lane * 8 + q;
    atomicAdd(&od[ch], xls[ch] * w);
  }
}

__global__ void k_ln(const float* __restrict__ agg, const float* __restrict__ h_in,
                     const float* __restrict__ gb, const float* __restrict__ g,
                     const float* __restrict__ bb, float* __restrict__ h_out) {
  __shared__ float sm[32];
  int n = blockIdx.x, c = threadIdx.x;
  float v = agg[(size_t)n * HDIM + c] + gb[c] + h_in[(size_t)n * HDIM + c];
  float mu  = blockReduceSum(v, sm) * (1.f / HDIM);
  float dx  = v - mu;
  float var = blockReduceSum(dx * dx, sm) * (1.f / HDIM);
  float y = dx * rsqrtf(var + 1e-5f) * g[c] + bb[c];
  h_out[(size_t)n * HDIM + c] = y > 0.f ? y : 0.f;
}

__global__ void k_graphrepr(const float* __restrict__ h, float* __restrict__ gr, int N) {
  int c = threadIdx.x;
  float s = 0.f, mx = -3.4e38f;
  for (int n = 0; n < N; ++n) {
    float v = h[(size_t)n * HDIM + c];
    s += v; mx = v > mx ? v : mx;
  }
  gr[c] = s / (float)N;
  gr[HDIM + c] = mx;
}

__global__ void k_cvt_w2(const float* __restrict__ W, _Float16* __restrict__ Wh, int n) {
  int t = blockIdx.x * blockDim.x + threadIdx.x;
  if (t < n) Wh[t] = (_Float16)W[t];
}

// block = one 8x8 (i,j) tile (tj>=ti), 128 threads = 4 waves.
// t[p][k] = relu(Ai[i]+Bj[j]) in LDS f16, then WMMA GEMM vs W2 [256x128],
// relu(+b2), dot w3, scatter to triu index.
__global__ void k_pairs(const float* __restrict__ Ai, const float* __restrict__ Bj,
                        const _Float16* __restrict__ W2, const float* __restrict__ b2,
                        const float* __restrict__ w3, const float* __restrict__ b3,
                        float* __restrict__ out, int N) {
  __shared__ _Float16 tile[64 * HDIM];      // 32 KB
  int T = N >> 3;
  int b = blockIdx.x, ti = 0;
  while (b >= T - ti) { b -= T - ti; ++ti; }
  int tj = ti + b;
  int i0 = ti * 8, j0 = tj * 8;
  int tid = threadIdx.x;
  {
    int p = tid >> 1, kh = (tid & 1) * 128;
    const float* ai = Ai + (size_t)(i0 + (p >> 3)) * HDIM + kh;
    const float* bj = Bj + (size_t)(j0 + (p & 7)) * HDIM + kh;
    _Float16* tr = tile + (size_t)p * HDIM + kh;
#pragma unroll 4
    for (int k = 0; k < 128; ++k) {
      float v = ai[k] + bj[k];
      tr[k] = (_Float16)(v > 0.f ? v : 0.f);
    }
  }
  __syncthreads();
  int lane = tid & 31, w = tid >> 5;
  int m0 = w * 16;                          // 16 pairs per wave
  v8f zero = {};
  v8f acc[8];
#pragma unroll
  for (int t = 0; t < 8; ++t) acc[t] = zero;
  for (int k0 = 0; k0 < HDIM; k0 += 32) {
    v16h a = load_A_lds(tile, m0, k0, lane);
    const _Float16* wp = W2 + (size_t)(k0 + ((lane >> 4) << 3)) * 128 + (lane & 15);
#pragma unroll
    for (int t = 0; t < 8; ++t) {
      v16h bf;
#pragma unroll
      for (int q = 0; q < 8; ++q) bf[q] = wp[q * 128 + t * 16];
#pragma unroll
      for (int q = 0; q < 8; ++q) bf[8 + q] = wp[(16 + q) * 128 + t * 16];
      acc[t] = __builtin_amdgcn_wmma_f32_16x16x32_f16(false, a, false, bf,
                                                      (short)0, acc[t], false, false);
    }
  }
  float s[8];
#pragma unroll
  for (int r = 0; r < 8; ++r) s[r] = 0.f;
  int nl = lane & 15;
#pragma unroll
  for (int t = 0; t < 8; ++t) {
    int n = t * 16 + nl;
    float b2v = b2[n], w3v = w3[n];
#pragma unroll
    for (int r = 0; r < 8; ++r) {
      float q = acc[t][r] + b2v;
      s[r] += (q > 0.f ? q : 0.f) * w3v;
    }
  }
#pragma unroll
  for (int r = 0; r < 8; ++r) {
    s[r] += __shfl_xor(s[r], 1, 32);
    s[r] += __shfl_xor(s[r], 2, 32);
    s[r] += __shfl_xor(s[r], 4, 32);
    s[r] += __shfl_xor(s[r], 8, 32);
  }
  float b3v = b3[0];
#pragma unroll
  for (int r = 0; r < 8; ++r) {
    if ((lane & 15) == r) {                 // lanes r and 16+r write
      int p = m0 + r + ((lane >> 4) << 3);
      int i = i0 + (p >> 3), j = j0 + (p & 7);
      if (i < j) {
        long long idx = (long long)i * (2LL * N - i - 1) / 2 + (j - i - 1);
        out[idx] = s[r] + b3v;
      }
    }
  }
}

__global__ void k_value(const float* __restrict__ gr, const float* __restrict__ W1,
                        const float* __restrict__ b1, const float* __restrict__ W2,
                        const float* __restrict__ b2, float* __restrict__ out) {
  __shared__ float sm[32];
  int j = threadIdx.x;
  float a = b1[j];
  for (int k = 0; k < 2 * HDIM; ++k) a += gr[k] * W1[(size_t)k * HDIM + j];
  a = a > 0.f ? a : 0.f;
  float tot = blockReduceSum(a * W2[j], sm);
  if (j == 0) out[0] = tot + b2[0];
}

__global__ void k_deg(const int* __restrict__ src, const int* __restrict__ dst,
                      const float* __restrict__ ea, float* __restrict__ deg1,
                      float* __restrict__ deg2, int E) {
  int e = blockIdx.x * blockDim.x + threadIdx.x;
  if (e >= E) return;
  float a0 = ea[(size_t)e * 3], a1 = ea[(size_t)e * 3 + 1], a2 = ea[(size_t)e * 3 + 2];
  int c = 0; float best = a0;
  if (a1 > best) { best = a1; c = 1; }
  if (a2 > best) { c = 2; }
  if (c == 1) { atomicAdd(&deg1[src[e]], 1.f); atomicAdd(&deg1[dst[e]], 1.f); }
  else if (c == 2) { atomicAdd(&deg2[src[e]], 1.f); atomicAdd(&deg2[dst[e]], 1.f); }
}

__global__ void k_energy(const float* __restrict__ deg1, const float* __restrict__ deg2,
                         const float* __restrict__ coupling, float* __restrict__ out, int N) {
  __shared__ float sm[32];
  float a = 0.f;
  for (int n = threadIdx.x; n < N; n += blockDim.x)
    a += deg1[n] * deg1[n] + deg2[n] * deg2[n];
  float tot = blockReduceSum(a, sm);
  if (threadIdx.x == 0) out[0] = tot / (2.f * (float)N) * coupling[0];
}

// ---------------------------------------------------------------- launch

extern "C" void kernel_launch(void* const* d_in, const int* in_sizes, int n_in,
                              void* d_out, int out_size, void* d_ws, size_t ws_size,
                              hipStream_t stream) {
  const int N = in_sizes[0];
  const int E = in_sizes[1] / 2;
  const int H = HDIM, L = 4;
  const long long TOT = (long long)E + N;
  const long long P = (long long)N * (N - 1) / 2;

  const float* x   = (const float*)d_in[0];
  const int*   src = (const int*)d_in[1];
  const int*   dst = src + E;
  const float* edge_attr = (const float*)d_in[2];
  const float* emb_W = (const float*)d_in[3];
  const float* emb_b = (const float*)d_in[4];
  const float* Wl = (const float*)d_in[5];
  const float* bl = (const float*)d_in[6];
  const float* Wr = (const float*)d_in[7];
  const float* br = (const float*)d_in[8];
  const float* We = (const float*)d_in[9];
  const float* att = (const float*)d_in[10];
  const float* gb = (const float*)d_in[11];
  const float* ln_g = (const float*)d_in[12];
  const float* ln_b = (const float*)d_in[13];
  const float* p1_W = (const float*)d_in[14];
  const float* p1_b = (const float*)d_in[15];
  const float* p2_W = (const float*)d_in[16];
  const float* p2_b = (const float*)d_in[17];
  const float* p3_W = (const float*)d_in[18];
  const float* p3_b = (const float*)d_in[19];
  const float* v1_W = (const float*)d_in[20];
  const float* v1_b = (const float*)d_in[21];
  const float* v2_W = (const float*)d_in[22];
  const float* v2_b = (const float*)d_in[23];
  const float* coupling = (const float*)d_in[24];
  float* out = (float*)d_out;

  char* cur = (char*)d_ws;
  auto carve = [&](size_t bytes) -> void* {
    void* p = (void*)cur;
    cur += (bytes + 255) & ~(size_t)255;
    return p;
  };
  float* hA      = (float*)carve((size_t)N * H * 4);
  float* hB      = (float*)carve((size_t)N * H * 4);
  float* xl      = (float*)carve((size_t)N * H * 4);
  float* xr      = (float*)carve((size_t)N * H * 4);
  float* alpha   = (float*)carve((size_t)TOT * NHEADS * 4);
  unsigned* amax = (unsigned*)carve((size_t)N * NHEADS * 4);
  float* denom   = (float*)carve((size_t)N * NHEADS * 4);
  float* aggbuf  = (float*)carve((size_t)N * H * 4);
  float* selfsum = (float*)carve((size_t)N * 3 * 4);
  float* selfcnt = (float*)carve((size_t)N * 4);
  float* selfatt = (float*)carve((size_t)N * 3 * 4);
  float* Ai      = (float*)carve((size_t)N * H * 4);
  float* Bj      = (float*)carve((size_t)N * H * 4);
  _Float16* W2h  = (_Float16*)carve((size_t)H * 128 * 2);
  float* grepr   = (float*)carve((size_t)2 * H * 4);
  float* deg1    = (float*)carve((size_t)N * 4);
  float* deg2    = (float*)carve((size_t)N * 4);

  auto cdiv = [](long long a, long long b) { return (int)((a + b - 1) / b); };

  // self-loop edge_attr fill (mean of incoming)
  hipMemsetAsync(selfsum, 0, (size_t)N * 3 * 4, stream);
  hipMemsetAsync(selfcnt, 0, (size_t)N * 4, stream);
  k_selfsum<<<cdiv(E, 256), 256, 0, stream>>>(src, dst, edge_attr, selfsum, selfcnt, E);
  k_selfattr<<<cdiv((long long)N * 3, 256), 256, 0, stream>>>(selfsum, selfcnt, selfatt, N);

  // embedding
  k_h0<<<N, H, 0, stream>>>(x, emb_W, emb_b, hA);

  float* h = hA;
  float* hn = hB;
  const int gemm_blocks = cdiv((long long)(N / 16) * (H / 16), 8); // 8 waves / block
  const int edge_wave_blocks = cdiv(TOT * 32, 256);

  for (int l = 0; l < L; ++l) {
    k_gemm_wmma<<<gemm_blocks, 256, 0, stream>>>(h, Wl + (size_t)l * H * H, bl + (size_t)l * H, xl, N, H, H);
    k_gemm_wmma<<<gemm_blocks, 256, 0, stream>>>(h, Wr + (size_t)l * H * H, br + (size_t)l * H, xr, N, H, H);
    hipMemsetAsync(amax, 0, (size_t)N * NHEADS * 4, stream);   // key 0 < all valid keys
    hipMemsetAsync(denom, 0, (size_t)N * NHEADS * 4, stream);
    k_alpha<<<edge_wave_blocks, 256, 0, stream>>>(xl, xr, src, dst, edge_attr, selfatt,
                                                  We + (size_t)l * 3 * H, att + (size_t)l * H,
                                                  alpha, amax, E, N);
    k_expsum<<<cdiv(TOT * NHEADS, 256), 256, 0, stream>>>(alpha, amax, denom, dst, E, N);
    hipMemsetAsync(aggbuf, 0, (size_t)N * H * 4, stream);
    k_agg<<<edge_wave_blocks, 256, 0, stream>>>(xl, alpha, denom, src, dst, aggbuf, E, N);
    k_ln<<<N, H, 0, stream>>>(aggbuf, h, gb + (size_t)l * H, ln_g + (size_t)l * H,
                              ln_b + (size_t)l * H, hn);
    float* t = h; h = hn; hn = t;
  }

  // graph representation
  k_graphrepr<<<1, H, 0, stream>>>(h, grepr, N);

  // pair head: Ai = h@W1_top + b1 ; Bj = h@W1_bot
  k_gemm_wmma<<<gemm_blocks, 256, 0, stream>>>(h, p1_W, p1_b, Ai, N, H, H);
  k_gemm_wmma<<<gemm_blocks, 256, 0, stream>>>(h, p1_W + (size_t)H * H, nullptr, Bj, N, H, H);
  k_cvt_w2<<<cdiv((long long)H * 128, 256), 256, 0, stream>>>(p2_W, W2h, H * 128);

  int T = N / 8;
  int pair_blocks = T * (T + 1) / 2;
  k_pairs<<<pair_blocks, 128, 0, stream>>>(Ai, Bj, W2h, p2_b, p3_W, p3_b, out, N);

  // value head
  k_value<<<1, H, 0, stream>>>(grepr, v1_W, v1_b, v2_W, v2_b, out + P);

  // Potts energy
  hipMemsetAsync(deg1, 0, (size_t)N * 4, stream);
  hipMemsetAsync(deg2, 0, (size_t)N * 4, stream);
  k_deg<<<cdiv(E, 256), 256, 0, stream>>>(src, dst, edge_attr, deg1, deg2, E);
  k_energy<<<1, 256, 0, stream>>>(deg1, deg2, coupling, out + P + 1, N);
}
